// TensorRingMatrix_89902255440661
// MI455X (gfx1250) — matrix operand order repaired and checked by reading, not compiled, not run on hardware
//
#include <hip/hip_runtime.h>

// ---------------------------------------------------------------------------
// TensorRing matrix:  out[b,j] = sum_k x[b,k] * W[j,k]
//   W[j,k] = trace(G0[:,a0,:]G1[:,a1,:]G2[:,a2,:]G3[:,a3,:]),
//   j = a0*32+a1, k = a2*32+a3
// Pipeline:
//   K1: P[j][p*8+q] = (G0G1)[p,q] ; Q[k][p*8+q] = (G2G3)[q,p]
//   K2: W[j,k] = dot64(P[j], Q[k]) -> split into bf16 hi/lo
//   K3: x -> bf16 hi/lo
//   K4: out = (xh+xl) @ (Wh+Wl)^T via v_wmma_f32_16x16x32_bf16,
//       3-term split (hh + hl + lh) for fp32-class accuracy.
//       Block = 8 wave32 = 128(M) x 64(N) tile; B staged in LDS (dbl-buffered)
//       so the 8 waves share W-strip loads.
// ---------------------------------------------------------------------------

#define IN_F  1024
#define OUT_F 1024
#define BATCH 4096

// LDS B-tile row stride in bf16 elements (64 cols x 32 K, padded 32->40)
#define BSTR 40

typedef __attribute__((ext_vector_type(16))) __bf16 v16bf;
typedef __attribute__((ext_vector_type(8)))  float  v8f;

union FragBF {
    v16bf v;
    uint4 q[2];
};

__device__ __forceinline__ unsigned short f32_to_bf16_rne(float f) {
    unsigned int u = __float_as_uint(f);
    unsigned int r = u + 0x7FFFu + ((u >> 16) & 1u);
    return (unsigned short)(r >> 16);
}
__device__ __forceinline__ float bf16_bits_to_f32(unsigned short h) {
    return __uint_as_float(((unsigned int)h) << 16);
}

// ---- Kernel 1: contract core pairs into P (1024x64) and Q (1024x64) --------
// G layout: (8, 32, 8) row-major -> G[r1*256 + a*8 + r2]
__global__ void tr_cores_kernel(const float* __restrict__ G0,
                                const float* __restrict__ G1,
                                const float* __restrict__ G2,
                                const float* __restrict__ G3,
                                float* __restrict__ P,
                                float* __restrict__ Q) {
    int blk = blockIdx.x;          // 0..2047
    int t   = threadIdx.x;         // 0..63 : t = p*8+q
    int p = t >> 3, q = t & 7;
    if (blk < 1024) {
        int a0 = blk >> 5, a1 = blk & 31;
        float s = 0.f;
#pragma unroll
        for (int b = 0; b < 8; ++b)
            s += G0[p * 256 + a0 * 8 + b] * G1[b * 256 + a1 * 8 + q];
        P[blk * 64 + t] = s;                       // (G0G1)[p,q]
    } else {
        int o = blk - 1024;
        int a2 = o >> 5, a3 = o & 31;
        float s = 0.f;
#pragma unroll
        for (int b = 0; b < 8; ++b)
            s += G2[q * 256 + a2 * 8 + b] * G3[b * 256 + a3 * 8 + p];
        Q[o * 64 + t] = s;                         // (G2G3)[q,p] at index p*8+q
    }
}

// ---- Kernel 2: W[j,k] = dot64(P[j],Q[k]); split to bf16 hi/lo --------------
__global__ __launch_bounds__(256) void tr_wbuild_kernel(
    const float* __restrict__ P, const float* __restrict__ Q,
    unsigned short* __restrict__ wh, unsigned short* __restrict__ wl) {
    __shared__ float Pt[16][64];
    __shared__ float Qt[16][64];
    int tx = threadIdx.x & 15, ty = threadIdx.x >> 4;
    int j0 = blockIdx.y * 16, k0 = blockIdx.x * 16;
#pragma unroll
    for (int r = 0; r < 4; ++r) {
        int idx = threadIdx.x + r * 256;           // 0..1023
        int row = idx >> 6, col = idx & 63;
        Pt[row][col] = P[(j0 + row) * 64 + col];
        Qt[row][col] = Q[(k0 + row) * 64 + col];
    }
    __syncthreads();
    float s = 0.f;
#pragma unroll
    for (int t = 0; t < 64; ++t) s += Pt[ty][t] * Qt[tx][t];
    unsigned short h = f32_to_bf16_rne(s);
    unsigned short l = f32_to_bf16_rne(s - bf16_bits_to_f32(h));
    int addr = (j0 + ty) * OUT_F + (k0 + tx);      // W row-major (j,k)
    wh[addr] = h;
    wl[addr] = l;
}

// ---- Kernel 3: split x into bf16 hi/lo -------------------------------------
__global__ void tr_xsplit_kernel(const float* __restrict__ x,
                                 unsigned short* __restrict__ xh,
                                 unsigned short* __restrict__ xl) {
    int i = blockIdx.x * blockDim.x + threadIdx.x;
    float v = x[i];
    unsigned short h = f32_to_bf16_rne(v);
    xh[i] = h;
    xl[i] = f32_to_bf16_rne(v - bf16_bits_to_f32(h));
}

// ---- Kernel 4: out = x @ W^T  (bf16 WMMA, 3-term fp32-split) ---------------
// Block (8 wave32) -> 128(M) x 64(N); wave -> 16 x 64; K stepped by 32.
// B tile (64 cols x 32 K, hi+lo) staged to LDS per k-step, double-buffered
// (K loop unrolled by 2 so buffer addresses are compile-time immediates).
// A frag (16x32 bf16): lane m = lane&15; lanes 0-15 hold K {kb..kb+7,kb+16..kb+23},
//   kb = k0 + 8*(lane>>4)  (ISA 16-bit A layout).
// B frag (32x16 bf16): lane n = lane&15; lanes 0-15 hold K {0..7,16..23} of the chunk,
//   lanes 16-31 hold K {8..15,24..31} (same order as A).
__global__ __launch_bounds__(256) void tr_gemm_kernel(
    const unsigned short* __restrict__ xh, const unsigned short* __restrict__ xl,
    const unsigned short* __restrict__ wh, const unsigned short* __restrict__ wl,
    float* __restrict__ out) {
    __shared__ unsigned short Bh[2][64 * BSTR];
    __shared__ unsigned short Bl[2][64 * BSTR];

    const int tid  = threadIdx.x;
    const int lane = tid & 31;
    const int wid  = tid >> 5;
    const int mBlock = blockIdx.x & 31;    // 32 blocks of 128 rows
    const int nBlock = blockIdx.x >> 5;    // 16 strips of 64 cols
    const int m0 = mBlock * 128 + wid * 16;
    const int n0 = nBlock * 64;

    // staging assignment: thread -> (col 0..63, k-part 0/8/16/24)
    const int scol = tid >> 2;
    const int skp  = (tid & 3) * 8;
    const unsigned short* wh_src = wh + (size_t)(n0 + scol) * OUT_F + skp;
    const unsigned short* wl_src = wl + (size_t)(n0 + scol) * OUT_F + skp;
    uint4* sBh0 = (uint4*)&Bh[0][scol * BSTR + skp];
    uint4* sBl0 = (uint4*)&Bl[0][scol * BSTR + skp];
    uint4* sBh1 = (uint4*)&Bh[1][scol * BSTR + skp];
    uint4* sBl1 = (uint4*)&Bl[1][scol * BSTR + skp];

    // A-fragment addressing
    const int arow  = m0 + (lane & 15);
    const int akOff = (lane >> 4) << 3;    // 0 or 8
    const unsigned short* a_h = xh + (size_t)arow * IN_F + akOff;
    const unsigned short* a_l = xl + (size_t)arow * IN_F + akOff;

    // B-fragment LDS addressing: col stride BSTR elems, halves 8 elems apart
    const int bbase = (lane & 15) * BSTR + ((lane >> 4) << 3);

    const v8f zero = {0.f, 0.f, 0.f, 0.f, 0.f, 0.f, 0.f, 0.f};
    v8f acc0 = zero, acc1 = zero, acc2 = zero, acc3 = zero;

    // prologue: stage k-chunk 0 into buffer 0
    *sBh0 = *(const uint4*)(wh_src + 0);
    *sBl0 = *(const uint4*)(wl_src + 0);
    __syncthreads();

#pragma unroll 2
    for (int ks = 0; ks < 32; ++ks) {
        const int k0  = ks * 32;
        const int buf = ks & 1;            // compile-time under unroll-by-2

        if (ks + 1 < 32) {                 // stage next chunk into other buffer
            const uint4 nh = *(const uint4*)(wh_src + k0 + 32);
            const uint4 nl = *(const uint4*)(wl_src + k0 + 32);
            if (buf) { *sBh0 = nh; *sBl0 = nl; }
            else     { *sBh1 = nh; *sBl1 = nl; }
        }

        FragBF ah, al;
        {
            const uint4* pah = (const uint4*)(a_h + k0);
            const uint4* pal = (const uint4*)(a_l + k0);
            ah.q[0] = pah[0]; ah.q[1] = pah[2];   // +16 elems = +2 uint4
            al.q[0] = pal[0]; al.q[1] = pal[2];
        }

        const unsigned short* __restrict__ bhp = buf ? &Bh[1][0] : &Bh[0][0];
        const unsigned short* __restrict__ blp = buf ? &Bl[1][0] : &Bl[0][0];

        FragBF bhf[4], blf[4];
#pragma unroll
        for (int nt = 0; nt < 4; ++nt) {
            const int off = bbase + nt * 16 * BSTR;
            bhf[nt].q[0] = *(const uint4*)&bhp[off];
            bhf[nt].q[1] = *(const uint4*)&bhp[off + 16];
            blf[nt].q[0] = *(const uint4*)&blp[off];
            blf[nt].q[1] = *(const uint4*)&blp[off + 16];
        }

        // hi*hi, hi*lo, lo*hi: dependent chains on each acc are 4 wmma apart
        acc0 = __builtin_amdgcn_wmma_f32_16x16x32_bf16(false, ah.v, false, bhf[0].v, (short)0, acc0, false, false);
        acc1 = __builtin_amdgcn_wmma_f32_16x16x32_bf16(false, ah.v, false, bhf[1].v, (short)0, acc1, false, false);
        acc2 = __builtin_amdgcn_wmma_f32_16x16x32_bf16(false, ah.v, false, bhf[2].v, (short)0, acc2, false, false);
        acc3 = __builtin_amdgcn_wmma_f32_16x16x32_bf16(false, ah.v, false, bhf[3].v, (short)0, acc3, false, false);

        acc0 = __builtin_amdgcn_wmma_f32_16x16x32_bf16(false, ah.v, false, blf[0].v, (short)0, acc0, false, false);
        acc1 = __builtin_amdgcn_wmma_f32_16x16x32_bf16(false, ah.v, false, blf[1].v, (short)0, acc1, false, false);
        acc2 = __builtin_amdgcn_wmma_f32_16x16x32_bf16(false, ah.v, false, blf[2].v, (short)0, acc2, false, false);
        acc3 = __builtin_amdgcn_wmma_f32_16x16x32_bf16(false, ah.v, false, blf[3].v, (short)0, acc3, false, false);

        acc0 = __builtin_amdgcn_wmma_f32_16x16x32_bf16(false, al.v, false, bhf[0].v, (short)0, acc0, false, false);
        acc1 = __builtin_amdgcn_wmma_f32_16x16x32_bf16(false, al.v, false, bhf[1].v, (short)0, acc1, false, false);
        acc2 = __builtin_amdgcn_wmma_f32_16x16x32_bf16(false, al.v, false, bhf[2].v, (short)0, acc2, false, false);
        acc3 = __builtin_amdgcn_wmma_f32_16x16x32_bf16(false, al.v, false, bhf[3].v, (short)0, acc3, false, false);

        __syncthreads();   // readers done with buf; next buf fully staged
    }

    // C/D layout: VGPR r, lanes 0-15 -> (M=r, N=lane), lanes 16-31 -> (M=r+8, N=lane-16)
    const int orow  = m0 + ((lane >> 4) << 3);
    const int ocol0 = n0 + (lane & 15);
    float* outp = out + (size_t)orow * OUT_F + ocol0;
#pragma unroll
    for (int r = 0; r < 8; ++r) outp[(size_t)r * OUT_F +  0] = acc0[r];
#pragma unroll
    for (int r = 0; r < 8; ++r) outp[(size_t)r * OUT_F + 16] = acc1[r];
#pragma unroll
    for (int r = 0; r < 8; ++r) outp[(size_t)r * OUT_F + 32] = acc2[r];
#pragma unroll
    for (int r = 0; r < 8; ++r) outp[(size_t)r * OUT_F + 48] = acc3[r];
}

// ---------------------------------------------------------------------------
extern "C" void kernel_launch(void* const* d_in, const int* in_sizes, int n_in,
                              void* d_out, int out_size, void* d_ws, size_t ws_size,
                              hipStream_t stream) {
    const float* x  = (const float*)d_in[0];
    const float* G0 = (const float*)d_in[1];
    const float* G1 = (const float*)d_in[2];
    const float* G2 = (const float*)d_in[3];
    const float* G3 = (const float*)d_in[4];
    float* out = (float*)d_out;

    char* ws = (char*)d_ws;
    float*          P  = (float*)(ws);                                       // 256 KB
    float*          Q  = (float*)(ws + (256u << 10));                        // 256 KB
    unsigned short* wh = (unsigned short*)(ws + (512u << 10));               // 2 MB
    unsigned short* wl = (unsigned short*)(ws + (512u << 10) + (2u << 20));  // 2 MB
    unsigned short* xh = (unsigned short*)(ws + (512u << 10) + (4u << 20));  // 8 MB
    unsigned short* xl = (unsigned short*)(ws + (512u << 10) + (12u << 20)); // 8 MB

    // K1: core pair contractions
    tr_cores_kernel<<<2048, 64, 0, stream>>>(G0, G1, G2, G3, P, Q);
    // K2: build W and split to bf16 hi/lo
    tr_wbuild_kernel<<<dim3(OUT_F / 16, IN_F / 16), 256, 0, stream>>>(P, Q, wh, wl);
    // K3: split x to bf16 hi/lo
    tr_xsplit_kernel<<<(BATCH * IN_F) / 256, 256, 0, stream>>>(x, xh, xl);
    // K4: main WMMA GEMM: 512 blocks x 8 wave32, 128x64 block tile
    tr_gemm_kernel<<<512, 256, 0, stream>>>(xh, xl, wh, wl, out);
}
